// SharedGNN_6691559047588
// MI455X (gfx1250) — matrix operand order repaired and checked by reading, not compiled, not run on hardware
//
#include <hip/hip_runtime.h>

#define SORT_N 65536

typedef __attribute__((ext_vector_type(16))) __bf16 v16bf;
typedef __attribute__((ext_vector_type(8)))  float  v8f;
typedef __attribute__((ext_vector_type(4)))  unsigned int u32x4;

// ---------- helpers ----------
__device__ __forceinline__ unsigned short f2bf(float f) {
  unsigned u = __float_as_uint(f);
  u += 0x7FFFu + ((u >> 16) & 1u);          // round-to-nearest-even
  return (unsigned short)(u >> 16);
}
union FRAG { v16bf f; u32x4 q[2]; };

// =====================================================================
// WMMA bf16 GEMM: Out[M x Nn] = (beta?Out:0) + A[M x Kdim]*W[Kdim x Nn] + bias
// Block tile 32 rows x 64 cols; 128 threads = 4 waves (wave32).
// LDS is staged in *fragment order* (CDNA5 ISA 7.12.2 lane/element layout
// applied at store time) so each A/B fragment is two ds_load_b128.
// Two v_wmma_f32_16x16x32_bf16 per wave per 32-wide K chunk.
// =====================================================================
__global__ void __launch_bounds__(128)
wmma_gemm_bf16(const float* __restrict__ A, int lda,
               const float* __restrict__ W, int ldw,
               const float* __restrict__ bias,
               float* __restrict__ Out, int ldo,
               int M, int Kdim, int beta)
{
  __shared__ __align__(16) unsigned short sAf[2][512];  // [rowTile][lane*16+i]
  __shared__ __align__(16) unsigned short sBf[4][512];  // [wave   ][lane*16+i]
  const int row0 = blockIdx.x * 32;
  const int col0 = blockIdx.y * 64;
  const int wave = threadIdx.x >> 5;
  const int lane = threadIdx.x & 31;
  const int g = lane >> 4, l = lane & 15;
  const int col = col0 + wave * 16 + l;

  v8f acc0, acc1;
#pragma unroll
  for (int j = 0; j < 8; ++j) {
    int r0 = row0 + g * 8 + j, r1 = r0 + 16;
    acc0[j] = (beta && r0 < M) ? Out[(long long)r0 * ldo + col] : 0.f;
    acc1[j] = (beta && r1 < M) ? Out[(long long)r1 * ldo + col] : 0.f;
  }

  for (int k0 = 0; k0 < Kdim; k0 += 32) {
    const bool interior = (row0 + 32 <= M) && (k0 + 32 <= Kdim);   // uniform
    // ---- stage A (32 rows x 32 k) in fragment order ----
    if (interior) {
#pragma unroll
      for (int t = threadIdx.x; t < 1024; t += 128) {
        int r = t >> 5, kk = t & 31;
        float v = A[(long long)(row0 + r) * lda + k0 + kk];
        int ag = (kk >> 3) & 1, ai = (kk & 7) | ((kk >> 4) << 3);
        sAf[r >> 4][(ag * 16 + (r & 15)) * 16 + ai] = f2bf(v);
      }
#pragma unroll
      for (int t = threadIdx.x; t < 2048; t += 128) {
        int kk = t >> 6, c = t & 63;
        float v = W[(long long)(k0 + kk) * ldw + col0 + c];
        sBf[c >> 4][((((kk >> 3) & 1) << 4) + (c & 15)) * 16 + ((kk & 7) | ((kk >> 4) << 3))] = f2bf(v);
      }
    } else {
#pragma unroll
      for (int t = threadIdx.x; t < 1024; t += 128) {
        int r = t >> 5, kk = t & 31;
        float v = (row0 + r < M && k0 + kk < Kdim)
                  ? A[(long long)(row0 + r) * lda + k0 + kk] : 0.f;
        int ag = (kk >> 3) & 1, ai = (kk & 7) | ((kk >> 4) << 3);
        sAf[r >> 4][(ag * 16 + (r & 15)) * 16 + ai] = f2bf(v);
      }
#pragma unroll
      for (int t = threadIdx.x; t < 2048; t += 128) {
        int kk = t >> 6, c = t & 63;
        float v = (k0 + kk < Kdim) ? W[(long long)(k0 + kk) * ldw + col0 + c] : 0.f;
        sBf[c >> 4][((((kk >> 3) & 1) << 4) + (c & 15)) * 16 + ((kk & 7) | ((kk >> 4) << 3))] = f2bf(v);
      }
    }
    __syncthreads();

    FRAG a0, a1, b;
    const u32x4* pA0 = (const u32x4*)&sAf[0][lane * 16];
    const u32x4* pA1 = (const u32x4*)&sAf[1][lane * 16];
    const u32x4* pB  = (const u32x4*)&sBf[wave][lane * 16];
    a0.q[0] = pA0[0]; a0.q[1] = pA0[1];
    a1.q[0] = pA1[0]; a1.q[1] = pA1[1];
    b.q[0]  = pB[0];  b.q[1]  = pB[1];

    acc0 = __builtin_amdgcn_wmma_f32_16x16x32_bf16(false, a0.f, false, b.f,
                                                   (short)0, acc0, false, false);
    acc1 = __builtin_amdgcn_wmma_f32_16x16x32_bf16(false, a1.f, false, b.f,
                                                   (short)0, acc1, false, false);
    __syncthreads();
  }

  float bv = bias ? bias[col] : 0.f;
#pragma unroll
  for (int j = 0; j < 8; ++j) {
    int r0 = row0 + g * 8 + j, r1 = r0 + 16;
    if (r0 < M) Out[(long long)r0 * ldo + col] = acc0[j] + bv;
    if (r1 < M) Out[(long long)r1 * ldo + col] = acc1[j] + bv;
  }
}

// ---------- elementwise / fill ----------
__global__ void fill_u32_kernel(unsigned* p, unsigned v, int n) {
  int i = blockIdx.x * 256 + threadIdx.x; if (i < n) p[i] = v;
}
__global__ void iota_kernel(int* p, int n) {
  int i = blockIdx.x * 256 + threadIdx.x; if (i < n) p[i] = i;
}
__global__ void bias_leaky_kernel(const float* __restrict__ acc, const float* __restrict__ bias,
                                  float* __restrict__ outp, int total, int cmask, float slope) {
  int i = blockIdx.x * 256 + threadIdx.x;
  if (i < total) { float v = acc[i] + bias[i & cmask]; outp[i] = v > 0.f ? v : slope * v; }
}
__global__ void residual_leaky_kernel(const float* __restrict__ hp, const float* __restrict__ T,
                                      float* __restrict__ outp, int total) {
  int i = blockIdx.x * 256 + threadIdx.x;
  if (i < total) { float v = hp[i] + T[i]; outp[i] = v > 0.f ? v : 0.01f * v; }
}

// ---------- GATv2 edge kernels ----------
template <int C>
__global__ void __launch_bounds__(256)
edge_score_kernel(const float* __restrict__ xl, const float* __restrict__ xr,
                  const float* __restrict__ eattr, const float* __restrict__ We,
                  const float* __restrict__ avec,
                  const int* __restrict__ src, const int* __restrict__ tgt,
                  float* __restrict__ s_out, unsigned* __restrict__ m_enc, int E)
{
  __shared__ float sWe[11 * C];
  __shared__ float sa[C];
  for (int t = threadIdx.x; t < 11 * C; t += 256) sWe[t] = We[t];
  for (int t = threadIdx.x; t < C; t += 256) sa[t] = avec[t];
  __syncthreads();
  int e = blockIdx.x * 256 + threadIdx.x;
  if (e >= E) return;
  const float* pl = xl + (long long)src[e] * C;
  const float* pr = xr + (long long)tgt[e] * C;
  __builtin_prefetch(pl, 0, 1);
  __builtin_prefetch(pr, 0, 1);
  float ea[11];
#pragma unroll
  for (int k = 0; k < 11; ++k) ea[k] = eattr[(long long)e * 11 + k];
  float s = 0.f;
  for (int c = 0; c < C; ++c) {
    float t = pl[c] + pr[c];
#pragma unroll
    for (int k = 0; k < 11; ++k) t += ea[k] * sWe[k * C + c];
    t = t > 0.f ? t : 0.2f * t;              // leaky_relu(0.2)
    s += t * sa[c];
  }
  s_out[e] = s;
  unsigned u = __float_as_uint(s);
  u = (u & 0x80000000u) ? ~u : (u | 0x80000000u);   // order-preserving encode
  atomicMax(&m_enc[tgt[e]], u);
}

__global__ void decode_m_kernel(const unsigned* __restrict__ me, float* __restrict__ mf, int n) {
  int i = blockIdx.x * 256 + threadIdx.x;
  if (i < n) {
    unsigned u = me[i];
    unsigned v = (u & 0x80000000u) ? (u ^ 0x80000000u) : ~u;
    float f = __uint_as_float(v);
    if ((v & 0x7F800000u) == 0x7F800000u) f = 0.f;   // non-finite -> 0
    mf[i] = f;
  }
}

__global__ void edge_soft_kernel(const float* __restrict__ s, const int* __restrict__ tgt,
                                 const float* __restrict__ m, float* __restrict__ al,
                                 float* __restrict__ den, int E) {
  int e = blockIdx.x * 256 + threadIdx.x;
  if (e < E) { int t = tgt[e]; float a = __expf(s[e] - m[t]); al[e] = a; atomicAdd(&den[t], a); }
}

template <int C>
__global__ void __launch_bounds__(256)
edge_agg_kernel(const float* __restrict__ al, const float* __restrict__ den,
                const float* __restrict__ xl,
                const int* __restrict__ src, const int* __restrict__ tgt,
                float* __restrict__ outp, int E)
{
  int e = blockIdx.x * 256 + threadIdx.x;
  if (e >= E) return;
  int t = tgt[e];
  const float* pl = xl + (long long)src[e] * C;
  __builtin_prefetch(pl, 0, 1);
  float coef = al[e] / (den[t] + 1e-16f);
  float* po = outp + (long long)t * C;
  for (int c = 0; c < C; ++c) atomicAdd(&po[c], coef * pl[c]);
}

// ---------- self-loop attr (fill_value='mean') ----------
__global__ void deg_attr_kernel(const int* __restrict__ tgt, const float* __restrict__ eattr,
                                float* __restrict__ deg, float* __restrict__ sat, int E) {
  int e = blockIdx.x * 256 + threadIdx.x;
  if (e >= E) return;
  int t = tgt[e];
  atomicAdd(&deg[t], 1.f);
  for (int k = 0; k < 11; ++k) atomicAdd(&sat[(long long)t * 11 + k], eattr[(long long)e * 11 + k]);
}
__global__ void norm_attr_kernel(float* __restrict__ sat, const float* __restrict__ deg, int N) {
  int i = blockIdx.x * 256 + threadIdx.x;
  if (i < N) { float d = fmaxf(deg[i], 1.f); for (int k = 0; k < 11; ++k) sat[(long long)i * 11 + k] /= d; }
}

// ---------- top-k ----------
__global__ void wnorm_kernel(const float* __restrict__ w, float* __restrict__ out) {
  __shared__ float red[256];
  int t = threadIdx.x; float v = w[t]; red[t] = v * v; __syncthreads();
  for (int s = 128; s > 0; s >>= 1) { if (t < s) red[t] += red[t + s]; __syncthreads(); }
  if (t == 0) out[0] = sqrtf(red[0]);
}
__global__ void score_key_kernel(const float* __restrict__ h2, const float* __restrict__ w,
                                 const float* __restrict__ normp, float* __restrict__ score,
                                 unsigned long long* __restrict__ keys, int N) {
  __shared__ float sw[256];
  sw[threadIdx.x] = w[threadIdx.x]; __syncthreads();
  int i = blockIdx.x * 256 + threadIdx.x;
  if (i < N) {
    const float* h = h2 + (long long)i * 256;
    float d = 0.f;
    for (int c = 0; c < 256; ++c) d += h[c] * sw[c];
    float s = tanhf(d / normp[0]);
    score[i] = s;
    unsigned u = __float_as_uint(s);
    u = (u & 0x80000000u) ? ~u : (u | 0x80000000u);
    u = ~u;                                   // descending sort via ascending keys
    keys[i] = ((unsigned long long)u << 32) | (unsigned)i;
  } else {
    keys[i] = ~0ull;                          // pad sorts to the end
  }
}
__global__ void bitonic_kernel(unsigned long long* d, unsigned j, unsigned k) {
  unsigned i = blockIdx.x * 256 + threadIdx.x;
  unsigned ixj = i ^ j;
  if (ixj > i) {
    unsigned long long a = d[i], b = d[ixj];
    bool asc = ((i & k) == 0u);
    if ((a > b) == asc) { d[i] = b; d[ixj] = a; }
  }
}
__global__ void build_hp_kernel(const unsigned long long* __restrict__ keys,
                                const float* __restrict__ score, const float* __restrict__ h2,
                                float* __restrict__ hp, int* __restrict__ perm, int K) {
  int kk = blockIdx.x, c = threadIdx.x;
  unsigned idx = (unsigned)(keys[kk] & 0xFFFFFFFFull);
  if (c == 0) perm[kk] = (int)idx;
  hp[(long long)kk * 256 + c] = h2[(long long)idx * 256 + c] * score[idx];
}
__global__ void set_newid_kernel(const int* __restrict__ perm, int* __restrict__ nid, int K) {
  int i = blockIdx.x * 256 + threadIdx.x; if (i < K) nid[perm[i]] = i;
}

// ---------- SAGE ----------
__global__ void sage_edge_kernel(const int* __restrict__ src, const int* __restrict__ tgt,
                                 const int* __restrict__ nid, const float* __restrict__ hp,
                                 float* __restrict__ agg, float* __restrict__ cnt, int E) {
  int e = blockIdx.x * 256 + threadIdx.x;
  if (e >= E) return;
  int ns = nid[src[e]], nt = nid[tgt[e]];
  if (ns < 0 || nt < 0) return;
  const float* ph = hp + (long long)ns * 256;
  __builtin_prefetch(ph, 0, 1);
  atomicAdd(&cnt[nt], 1.f);
  float* pa = agg + (long long)nt * 256;
  for (int c = 0; c < 256; ++c) atomicAdd(&pa[c], ph[c]);
}
__global__ void mean_kernel(float* __restrict__ agg, const float* __restrict__ cnt, int K) {
  long long i = (long long)blockIdx.x * 256 + threadIdx.x;
  agg[i] /= fmaxf(cnt[blockIdx.x], 1.f);
}

// =====================================================================
extern "C" void kernel_launch(void* const* d_in, const int* in_sizes, int n_in,
                              void* d_out, int out_size, void* d_ws, size_t ws_size,
                              hipStream_t stream)
{
  const float* x    = (const float*)d_in[0];
  const int*   ei   = (const int*)  d_in[1];
  const float* ea   = (const float*)d_in[2];
  const float* W1l  = (const float*)d_in[3];
  const float* b1l  = (const float*)d_in[4];
  const float* W1r  = (const float*)d_in[5];
  const float* b1r  = (const float*)d_in[6];
  const float* W1e  = (const float*)d_in[7];
  const float* a1   = (const float*)d_in[8];
  const float* bias1= (const float*)d_in[9];
  const float* W2l  = (const float*)d_in[10];
  const float* b2l  = (const float*)d_in[11];
  const float* W2r  = (const float*)d_in[12];
  const float* b2r  = (const float*)d_in[13];
  const float* W2e  = (const float*)d_in[14];
  const float* a2   = (const float*)d_in[15];
  const float* bias2= (const float*)d_in[16];
  const float* pw   = (const float*)d_in[17];
  const float* Wsl  = (const float*)d_in[18];
  const float* bsl  = (const float*)d_in[19];
  const float* Wsr  = (const float*)d_in[20];
  float* out = (float*)d_out;
  (void)n_in; (void)out_size; (void)ws_size;

  const int N = in_sizes[0] / 30;
  const int E = in_sizes[1] / 2;
  const int K = (int)((4LL * N + 4) / 5);     // ceil(0.8*N)
  const int* src = ei;
  const int* tgt = ei + E;

  float* ws = (float*)d_ws;
  size_t off = 0;
  auto alloc = [&](size_t n) { float* p = ws + off; off += (n + 1) & ~(size_t)1; return p; };
  float* A   = alloc((size_t)N * 256);        // xl1/xl2, later hp
  float* Bb  = alloc((size_t)N * 256);        // xr1/xr2, later agg/mean
  float* Cc  = alloc((size_t)N * 256);        // accum -> h2, later y
  float* D   = alloc((size_t)N * 64);         // h1
  float* S1  = alloc((size_t)E);              // edge scores
  float* S2  = alloc((size_t)E);              // edge alphas
  unsigned long long* keys = (unsigned long long*)alloc(2 * (size_t)SORT_N);
  unsigned* m_enc = (unsigned*)alloc(N);
  float* mF  = alloc(N);                      // decoded max; later ||w|| scalar
  float* den = alloc(N);
  float* deg = alloc(N);
  float* sat = alloc((size_t)N * 11);
  float* ssf = alloc(N);                      // self-loop scores
  float* asf = alloc(N);                      // self-loop alphas
  float* score = alloc(N);
  int* perm = (int*)alloc(K);
  int* nid  = (int*)alloc(N);
  float* cnt = alloc(K);
  int* iot  = (int*)alloc(N);

  const unsigned ENC_NEG_INF = 0x007FFFFFu;   // encode(-inf)
  dim3 blk(256);
  auto g1 = [](int n) { return dim3((unsigned)((n + 255) / 256)); };

  iota_kernel<<<g1(N), blk, 0, stream>>>(iot, N);

  // ---- conv1: xl1 = x@W1l+b1l, xr1 = x@W1r+b1r  (WMMA) ----
  {
    dim3 grid((N + 31) / 32, 1);
    wmma_gemm_bf16<<<grid, 128, 0, stream>>>(x, 30, W1l, 64, b1l, A, 64, N, 30, 0);
    wmma_gemm_bf16<<<grid, 128, 0, stream>>>(x, 30, W1r, 64, b1r, Bb, 64, N, 30, 0);
  }
  fill_u32_kernel<<<g1(N), blk, 0, stream>>>(m_enc, ENC_NEG_INF, N);
  fill_u32_kernel<<<g1(N), blk, 0, stream>>>((unsigned*)den, 0u, N);
  edge_score_kernel<64><<<g1(E), blk, 0, stream>>>(A, Bb, ea, W1e, a1, src, tgt, S1, m_enc, E);
  decode_m_kernel<<<g1(N), blk, 0, stream>>>(m_enc, mF, N);
  edge_soft_kernel<<<g1(E), blk, 0, stream>>>(S1, tgt, mF, S2, den, E);
  fill_u32_kernel<<<g1(N * 64), blk, 0, stream>>>((unsigned*)Cc, 0u, N * 64);
  edge_agg_kernel<64><<<g1(E), blk, 0, stream>>>(S2, den, A, src, tgt, Cc, E);
  bias_leaky_kernel<<<g1(N * 64), blk, 0, stream>>>(Cc, bias1, D, N * 64, 63, 0.01f);

  // ---- self-loop edge_attr = mean of incoming ----
  fill_u32_kernel<<<g1(N), blk, 0, stream>>>((unsigned*)deg, 0u, N);
  fill_u32_kernel<<<g1(N * 11), blk, 0, stream>>>((unsigned*)sat, 0u, N * 11);
  deg_attr_kernel<<<g1(E), blk, 0, stream>>>(tgt, ea, deg, sat, E);
  norm_attr_kernel<<<g1(N), blk, 0, stream>>>(sat, deg, N);

  // ---- conv2: xl2/xr2 (WMMA), edges + self loops ----
  {
    dim3 grid((N + 31) / 32, 4);
    wmma_gemm_bf16<<<grid, 128, 0, stream>>>(D, 64, W2l, 256, b2l, A, 256, N, 64, 0);
    wmma_gemm_bf16<<<grid, 128, 0, stream>>>(D, 64, W2r, 256, b2r, Bb, 256, N, 64, 0);
  }
  fill_u32_kernel<<<g1(N), blk, 0, stream>>>(m_enc, ENC_NEG_INF, N);
  fill_u32_kernel<<<g1(N), blk, 0, stream>>>((unsigned*)den, 0u, N);
  edge_score_kernel<256><<<g1(E), blk, 0, stream>>>(A, Bb, ea, W2e, a2, src, tgt, S1, m_enc, E);
  edge_score_kernel<256><<<g1(N), blk, 0, stream>>>(A, Bb, sat, W2e, a2, iot, iot, ssf, m_enc, N);
  decode_m_kernel<<<g1(N), blk, 0, stream>>>(m_enc, mF, N);
  edge_soft_kernel<<<g1(E), blk, 0, stream>>>(S1, tgt, mF, S2, den, E);
  edge_soft_kernel<<<g1(N), blk, 0, stream>>>(ssf, iot, mF, asf, den, N);
  fill_u32_kernel<<<g1(N * 256), blk, 0, stream>>>((unsigned*)Cc, 0u, N * 256);
  edge_agg_kernel<256><<<g1(E), blk, 0, stream>>>(S2, den, A, src, tgt, Cc, E);
  edge_agg_kernel<256><<<g1(N), blk, 0, stream>>>(asf, den, A, iot, iot, Cc, N);
  bias_leaky_kernel<<<g1(N * 256), blk, 0, stream>>>(Cc, bias2, Cc, N * 256, 255, 0.01f);

  // ---- TopKPooling: score, bitonic sort 65536 u64 keys, gather ----
  wnorm_kernel<<<1, 256, 0, stream>>>(pw, mF);
  score_key_kernel<<<SORT_N / 256, 256, 0, stream>>>(Cc, pw, mF, score, keys, N);
  for (unsigned kk = 2; kk <= (unsigned)SORT_N; kk <<= 1)
    for (unsigned j = kk >> 1; j >= 1; j >>= 1)
      bitonic_kernel<<<SORT_N / 256, 256, 0, stream>>>(keys, j, kk);
  build_hp_kernel<<<K, 256, 0, stream>>>(keys, score, Cc, A, perm, K);
  fill_u32_kernel<<<g1(N), blk, 0, stream>>>((unsigned*)nid, 0xFFFFFFFFu, N);
  set_newid_kernel<<<g1(K), blk, 0, stream>>>(perm, nid, K);

  // ---- SAGE: mean-agg + two WMMA GEMMs + residual leaky ----
  fill_u32_kernel<<<g1(K * 256), blk, 0, stream>>>((unsigned*)Bb, 0u, K * 256);
  fill_u32_kernel<<<g1(K), blk, 0, stream>>>((unsigned*)cnt, 0u, K);
  sage_edge_kernel<<<g1(E), blk, 0, stream>>>(src, tgt, nid, A, Bb, cnt, E);
  mean_kernel<<<K, 256, 0, stream>>>(Bb, cnt, K);
  {
    dim3 grid((K + 31) / 32, 4);
    wmma_gemm_bf16<<<grid, 128, 0, stream>>>(Bb, 256, Wsl, 256, bsl, Cc, 256, K, 256, 0);
    wmma_gemm_bf16<<<grid, 128, 0, stream>>>(A, 256, Wsr, 256, (const float*)nullptr, Cc, 256, K, 256, 1);
  }
  residual_leaky_kernel<<<g1(K * 256), blk, 0, stream>>>(A, Cc, out, K * 256);
}
